// QKVAttention_47614007443764
// MI455X (gfx1250) — matrix operand order repaired and checked by reading, not compiled, not run on hardware
//
#include <hip/hip_runtime.h>

// ---------------------------------------------------------------------------
// QKVAttention for MI455X (gfx1250): bf16 WMMA flash-attention + algebraic
// reassociation of the no-softmax branch (attn_st = (V K^T) Q).
// ---------------------------------------------------------------------------

typedef __attribute__((ext_vector_type(16))) __bf16 v16bf;
typedef __attribute__((ext_vector_type(8)))  __bf16 v8bf;
typedef __attribute__((ext_vector_type(4)))  __bf16 v4bf;
typedef __attribute__((ext_vector_type(8)))  float  v8f;
typedef __attribute__((ext_vector_type(4)))  float  v4f;
typedef __attribute__((ext_vector_type(4)))  int    v4i;

static __device__ __forceinline__ v8f wmma_bf16f32(v16bf a, v16bf b, v8f c) {
  // D = A(16x32 bf16) * B(32x16 bf16) + C(16x16 f32)
  return __builtin_amdgcn_wmma_f32_16x16x32_bf16(false, a, false, b,
                                                 (short)0, c, false, false);
}

#if defined(__has_builtin)
#if __has_builtin(__builtin_amdgcn_global_load_async_to_lds_b128) && \
    __has_builtin(__builtin_amdgcn_s_wait_asynccnt)
#define USE_ASYNC_LDS 1
#endif
#endif

#define T_IMG 4096
#define S_ST  1024
#define WIDTH 1536
#define SCALE 0.3535533905932738f   // 1/sqrt(sqrt(64))

// ---------------------------------------------------------------------------
// Kernel 1: f32 -> bf16 conversion, folding the q/k scale (rows < 1024 of the
// width-1536 dim are q and k sections; v section unscaled).
// ---------------------------------------------------------------------------
__global__ __launch_bounds__(256) void cvt_scale_kernel(
    const float* __restrict__ in, __bf16* __restrict__ out, int rowlen) {
  size_t i = ((size_t)blockIdx.x * 256 + threadIdx.x) * 4;
  v4f v = *(const v4f*)(in + i);
  size_t row = (i / (size_t)rowlen) % WIDTH;
  float s = (row < 1024) ? SCALE : 1.0f;
  v4bf o;
#pragma unroll
  for (int j = 0; j < 4; ++j) o[j] = (__bf16)(v[j] * s);
  *(v4bf*)(out + i) = o;
}

// ---------------------------------------------------------------------------
// Kernel 2: flash attention, image queries over st K/V (with softmax).
// Block = 256 threads (8 waves); block handles one head x 128 t; wave = 16 t.
// Logits computed as W^T = K*Q so softmax stats live in a lane + its ^16
// partner, O-rescale is a per-lane scalar, and P(C-frag)->B-frag is one
// shfl_xor(16) per element pair. K chunk is staged transposed in LDS
// (ds_store_b16 path, shared by all 8 waves); V chunk is staged untransposed
// via GLOBAL_LOAD_ASYNC_TO_LDS_B128 (ASYNCcnt path) when available.
// ---------------------------------------------------------------------------
__global__ __launch_bounds__(256, 1) void attn_softmax_kernel(
    const __bf16* __restrict__ img, const __bf16* __restrict__ st,
    float* __restrict__ out) {
  const int head  = blockIdx.x >> 5;  // 0..15 = b*8 + h
  const int ttile = blockIdx.x & 31;  // 32 tiles of 128 t
  const int b = head >> 3, h = head & 7;

  const __bf16* Q = img + ((size_t)b * WIDTH + 0    + h * 64) * T_IMG;
  const __bf16* K = st  + ((size_t)b * WIDTH + 512  + h * 64) * S_ST;
  const __bf16* V = st  + ((size_t)b * WIDTH + 1024 + h * 64) * S_ST;
  float* O = out + ((size_t)b * 512 + h * 64) * T_IMG;

  const int wave = threadIdx.x >> 5;
  const int lane = threadIdx.x & 31;
  const int ln16 = lane & 15;
  const int hi   = lane >> 4;  // 0/1
  const int t0   = ttile * 128 + wave * 16 + ln16;  // this lane's t column

  __shared__ __align__(16) __bf16 Kt[32][72];  // [s_local][c], stride 144B
  __shared__ __align__(16) __bf16 Vs[64][32];  // [c][s_local], stride 64B

  // Q B-fragments, hoisted: element e -> c = cs*32 + e + 16*hi, col = t0.
  v16bf qf[2];
#pragma unroll
  for (int cs = 0; cs < 2; ++cs) {
    const int cbase = cs * 32 + 16 * hi;
#pragma unroll
    for (int e = 0; e < 16; ++e)
      qf[cs][e] = Q[(size_t)(cbase + e) * T_IMG + t0];
  }

  float m_run = -3.0e38f, l_run = 0.0f;
  v8f acc[4] = {};  // O accumulators, rows c = g*16 + r + 8*hi, col t0

  for (int sb = 0; sb < S_ST; sb += 32) {
    __syncthreads();
    // ---- Stage K chunk (64c x 32s) transposed into LDS, coalesced. ----
    {
      const int sloc = (threadIdx.x & 7) * 4;
#pragma unroll
      for (int p = 0; p < 2; ++p) {
        const int c = p * 32 + (threadIdx.x >> 3);
        const __bf16* src = K + (size_t)c * S_ST + sb + sloc;
        if (sb + 32 < S_ST) __builtin_prefetch(src + 32, 0, 1);
        v4bf kv = *(const v4bf*)src;
#pragma unroll
        for (int j = 0; j < 4; ++j) Kt[sloc + j][c] = kv[j];
      }
    }
    // ---- Stage V chunk (64c x 32s), one b128 per thread, async->LDS. ----
    {
      const int c = threadIdx.x >> 1;
      const int so8 = (threadIdx.x & 1) * 8;
      const __bf16* gsrc = V + (size_t)c * S_ST + sb + so8;
      __bf16* ldst = &Vs[c][so8];
#if defined(USE_ASYNC_LDS)
      __builtin_amdgcn_global_load_async_to_lds_b128(
          (__attribute__((address_space(1))) v4i*)gsrc,
          (__attribute__((address_space(3))) v4i*)ldst, 0, 0);
      __builtin_amdgcn_s_wait_asynccnt(0);
#else
      *(v8bf*)ldst = *(const v8bf*)gsrc;
#endif
    }
    __syncthreads();

    // ---- A(K) fragments from LDS: row s = stile*16 + ln16,
    //      e -> c = cs*32 + 16*hi + e. Two b128 reads. ----
    v16bf kf[2][2];
#pragma unroll
    for (int stile = 0; stile < 2; ++stile) {
      const int srow = stile * 16 + ln16;
#pragma unroll
      for (int cs = 0; cs < 2; ++cs) {
        const int cb = cs * 32 + 16 * hi;
        v8bf lo = *(const v8bf*)&Kt[srow][cb];
        v8bf hi8 = *(const v8bf*)&Kt[srow][cb + 8];
#pragma unroll
        for (int j = 0; j < 8; ++j) {
          kf[stile][cs][j] = lo[j];
          kf[stile][cs][8 + j] = hi8[j];
        }
      }
    }

    // ---- Logits W^T (two 16s x 16t tiles), K-dim c=64 -> 2 WMMAs each. ----
    v8f c0 = {}, c1 = {};
    c0 = wmma_bf16f32(kf[0][0], qf[0], c0);
    c0 = wmma_bf16f32(kf[0][1], qf[1], c0);
    c1 = wmma_bf16f32(kf[1][0], qf[0], c1);
    c1 = wmma_bf16f32(kf[1][1], qf[1], c1);

    // ---- Online softmax over this 32-s chunk (per t = per lane pair). ----
    float mx = c0[0];
#pragma unroll
    for (int j = 0; j < 8; ++j) {
      mx = fmaxf(mx, c0[j]);
      mx = fmaxf(mx, c1[j]);
    }
    mx = fmaxf(mx, __shfl_xor(mx, 16, 32));
    const float m_new = fmaxf(m_run, mx);
    const float corr = __expf(m_run - m_new);
    float p0[8], p1[8], ps = 0.0f;
#pragma unroll
    for (int j = 0; j < 8; ++j) {
      p0[j] = __expf(c0[j] - m_new);
      p1[j] = __expf(c1[j] - m_new);
      ps += p0[j] + p1[j];
    }
    ps += __shfl_xor(ps, 16, 32);
    l_run = l_run * corr + ps;
    m_run = m_new;
#pragma unroll
    for (int g = 0; g < 4; ++g) acc[g] *= corr;

    // ---- P C-frag -> B-frag (k = s_local = e + 16*hi) via xor-16. ----
    v16bf pf;
#pragma unroll
    for (int j = 0; j < 8; ++j) {
      const float a = __shfl_xor(p0[j], 16, 32);   // partner's c0 row j
      const float bsw = __shfl_xor(p1[j], 16, 32); // partner's c1 row j
      const float e_lo = hi ? bsw : p0[j];  // element j
      const float e_hi = hi ? p1[j] : a;    // element j+8
      pf[j] = (__bf16)e_lo;
      pf[8 + j] = (__bf16)e_hi;
    }

    // ---- O += V(16c x 32s as A) * P(32s x 16t as B), 4 c-groups. ----
#pragma unroll
    for (int g = 0; g < 4; ++g) {
      const __bf16* vp = &Vs[g * 16 + ln16][16 * hi];
      v8bf vlo = *(const v8bf*)vp;          // s_local = 16*hi + 0..7
      v8bf vhi = *(const v8bf*)(vp + 8);    // s_local = 16*hi + 8..15
      v16bf vf;
#pragma unroll
      for (int j = 0; j < 8; ++j) {
        vf[j] = vlo[j];
        vf[8 + j] = vhi[j];
      }
      acc[g] = wmma_bf16f32(vf, pf, acc[g]);
    }
  }

  const float inv = 1.0f / l_run;
#pragma unroll
  for (int g = 0; g < 4; ++g)
#pragma unroll
    for (int r = 0; r < 8; ++r) {
      const int c = g * 16 + r + 8 * hi;
      O[(size_t)c * T_IMG + t0] = acc[g][r] * inv;
    }
}

// ---------------------------------------------------------------------------
// Kernel 3a: M[b,h] = V_img * K_img^T  (64x64, K-dim s = 4096).
// One block per head; 8 waves split s; ds_add_f32 reduction in LDS.
// ---------------------------------------------------------------------------
__global__ __launch_bounds__(256, 1) void vk_outer_kernel(
    const __bf16* __restrict__ img, float* __restrict__ Mout) {
  const int head = blockIdx.x;
  const int b = head >> 3, h = head & 7;
  const __bf16* K = img + ((size_t)b * WIDTH + 512  + h * 64) * T_IMG;
  const __bf16* V = img + ((size_t)b * WIDTH + 1024 + h * 64) * T_IMG;

  __shared__ float Ms[64 * 64];
  for (int i = threadIdx.x; i < 4096; i += 256) Ms[i] = 0.0f;
  __syncthreads();

  const int wave = threadIdx.x >> 5;
  const int lane = threadIdx.x & 31;
  const int ln16 = lane & 15;
  const int hi   = lane >> 4;

  v8f acc[4][4] = {};  // [c-group][c'-group]
  const int s_end = wave * 512 + 512;
  for (int sb = wave * 512; sb < s_end; sb += 32) {
    v16bf af[4], bfr[4];
#pragma unroll
    for (int g = 0; g < 4; ++g) {
      // A = V: row c = g*16+ln16, e -> s = sb + e + 16*hi (contig 16)
      const __bf16* vp = V + (size_t)(g * 16 + ln16) * T_IMG + sb + 16 * hi;
      v8bf vlo = *(const v8bf*)vp;
      v8bf vhi = *(const v8bf*)(vp + 8);
      // B = K^T: col c' = g*16+ln16, e -> s = sb + e + 16*hi (contig 16)
      const __bf16* kp = K + (size_t)(g * 16 + ln16) * T_IMG + sb + 16 * hi;
      v8bf klo = *(const v8bf*)kp;
      v8bf khi = *(const v8bf*)(kp + 8);
#pragma unroll
      for (int j = 0; j < 8; ++j) {
        af[g][j] = vlo[j]; af[g][8 + j] = vhi[j];
        bfr[g][j] = klo[j]; bfr[g][8 + j] = khi[j];
      }
    }
#pragma unroll
    for (int cg = 0; cg < 4; ++cg)
#pragma unroll
      for (int pg = 0; pg < 4; ++pg)
        acc[cg][pg] = wmma_bf16f32(af[cg], bfr[pg], acc[cg][pg]);
  }

#pragma unroll
  for (int cg = 0; cg < 4; ++cg)
#pragma unroll
    for (int pg = 0; pg < 4; ++pg)
#pragma unroll
      for (int r = 0; r < 8; ++r) {
        const int c = cg * 16 + r + 8 * hi;
        const int cp = pg * 16 + ln16;
        atomicAdd(&Ms[c * 64 + cp], acc[cg][pg][r]);  // ds_add_f32
      }
  __syncthreads();
  float* Mg = Mout + (size_t)head * 4096;
  for (int i = threadIdx.x; i < 4096; i += 256) Mg[i] = Ms[i];
}

// ---------------------------------------------------------------------------
// Kernel 3b: attn_st = M * Q_st  (64 x 1024 per head, K-dim c' = 64).
// ---------------------------------------------------------------------------
__global__ __launch_bounds__(256, 1) void mq_gemm_kernel(
    const float* __restrict__ Mg, const __bf16* __restrict__ st,
    float* __restrict__ out_st) {
  const int head = blockIdx.x;
  const int b = head >> 3, h = head & 7;
  const float* M = Mg + (size_t)head * 4096;
  const __bf16* Q = st + ((size_t)b * WIDTH + h * 64) * S_ST;
  float* O = out_st + ((size_t)b * 512 + h * 64) * S_ST;

  const int wave = threadIdx.x >> 5;
  const int lane = threadIdx.x & 31;
  const int ln16 = lane & 15;
  const int hi   = lane >> 4;

  // A = M fragments: row c = cg*16+ln16, e -> c' = cs*32+16*hi+e
  v16bf af[4][2];
#pragma unroll
  for (int cg = 0; cg < 4; ++cg)
#pragma unroll
    for (int cs = 0; cs < 2; ++cs) {
      const float* mp = M + (cg * 16 + ln16) * 64 + cs * 32 + 16 * hi;
#pragma unroll
      for (int j = 0; j < 8; ++j) {
        af[cg][cs][j] = (__bf16)mp[j];
        af[cg][cs][8 + j] = (__bf16)mp[8 + j];
      }
    }

  const int t_end = wave * 128 + 128;
  for (int tt = wave * 128; tt < t_end; tt += 16) {
    const int t0 = tt + ln16;
    v16bf qf[2];
#pragma unroll
    for (int cs = 0; cs < 2; ++cs) {
      const int cbase = cs * 32 + 16 * hi;
#pragma unroll
      for (int e = 0; e < 16; ++e)
        qf[cs][e] = Q[(size_t)(cbase + e) * S_ST + t0];
    }
#pragma unroll
    for (int cg = 0; cg < 4; ++cg) {
      v8f a = {};
      a = wmma_bf16f32(af[cg][0], qf[0], a);
      a = wmma_bf16f32(af[cg][1], qf[1], a);
#pragma unroll
      for (int r = 0; r < 8; ++r)
        O[(size_t)(cg * 16 + r + 8 * hi) * S_ST + t0] = a[r];
    }
  }
}

// ---------------------------------------------------------------------------
extern "C" void kernel_launch(void* const* d_in, const int* in_sizes, int n_in,
                              void* d_out, int out_size, void* d_ws,
                              size_t ws_size, hipStream_t stream) {
  const float* img_f = (const float*)d_in[0];  // (2,1536,4096)
  const float* st_f  = (const float*)d_in[1];  // (2,1536,1024)
  float* out = (float*)d_out;

  const size_t IMG_ELEMS = (size_t)2 * WIDTH * T_IMG;  // 12,582,912
  const size_t ST_ELEMS  = (size_t)2 * WIDTH * S_ST;   //  3,145,728

  char* ws = (char*)d_ws;
  __bf16* img_bf = (__bf16*)ws;
  __bf16* st_bf  = (__bf16*)(ws + IMG_ELEMS * 2);
  float*  Mbuf   = (float*)(ws + (IMG_ELEMS + ST_ELEMS) * 2);

  // 1) convert + fold q/k scale into bf16 copies
  cvt_scale_kernel<<<(int)(IMG_ELEMS / 4 / 256), 256, 0, stream>>>(
      img_f, img_bf, T_IMG);
  cvt_scale_kernel<<<(int)(ST_ELEMS / 4 / 256), 256, 0, stream>>>(
      st_f, st_bf, S_ST);

  // 2) attn_i: flash attention (softmax), 16 heads x 32 t-tiles
  attn_softmax_kernel<<<512, 256, 0, stream>>>(img_bf, st_bf, out);

  // 3) attn_st (no softmax) = (V_img K_img^T) Q_st
  vk_outer_kernel<<<16, 256, 0, stream>>>(img_bf, Mbuf);
  mq_gemm_kernel<<<16, 256, 0, stream>>>(
      Mbuf, st_bf, out + (size_t)2 * 512 * T_IMG);
}
